// PVConv_8624294330484
// MI455X (gfx1250) — matrix operand order repaired and checked by reading, not compiled, not run on hardware
//
#include <hip/hip_runtime.h>
#include <hip/hip_bf16.h>

typedef __attribute__((ext_vector_type(16))) _Float16 v16h;
typedef __attribute__((ext_vector_type(8)))  float    v8f;

#define RES    32
#define RC     32768   // RES^3
#define NPTS   32768
#define BATCH  8
#define CH     64
#define BN_EPS 1e-4f

union F16x16 { v16h v; uint4 u[2]; };

// tap offset tables: tap = (di+1)*9 + (dj+1)*3 + (dk+1)
__device__ __constant__ int C_DI[27] = {-1,-1,-1,-1,-1,-1,-1,-1,-1, 0,0,0,0,0,0,0,0,0, 1,1,1,1,1,1,1,1,1};
__device__ __constant__ int C_DJ[27] = {-1,-1,-1, 0,0,0, 1,1,1, -1,-1,-1, 0,0,0, 1,1,1, -1,-1,-1, 0,0,0, 1,1,1};
__device__ __constant__ int C_DK[27] = {-1,0,1, -1,0,1, -1,0,1, -1,0,1, -1,0,1, -1,0,1, -1,0,1, -1,0,1, -1,0,1};

// ---- CDNA5 async global->LDS copy (ASYNCcnt-tracked), 16B per lane
__device__ __forceinline__ unsigned lds_off(const void* p) {
    // generic LDS pointer: low 32 bits are the wave-relative LDS byte offset
    return (unsigned)(uintptr_t)p;
}
__device__ __forceinline__ void async_copy_b128(unsigned dst_lds, const void* src_global) {
    asm volatile("global_load_async_to_lds_b128 %0, %1, off"
                 :: "v"(dst_lds), "v"(src_global) : "memory");
}
__device__ __forceinline__ void wait_async0() {
    asm volatile("s_wait_asynccnt 0" ::: "memory");
}

// ---------------------------------------------------------------- utilities
__global__ void k_zero_f32(float* p, int n) {
    int i = blockIdx.x * blockDim.x + threadIdx.x;
    if (i < n) p[i] = 0.0f;
}

__global__ void k_coord_mean(const float* __restrict__ coords, float* __restrict__ mean) {
    int ba = blockIdx.x;                 // b*3+axis
    const float* src = coords + (size_t)ba * NPTS;
    __shared__ float red[256];
    float s = 0.f;
    for (int i = threadIdx.x; i < NPTS; i += 256) s += src[i];
    red[threadIdx.x] = s; __syncthreads();
    for (int st = 128; st > 0; st >>= 1) {
        if (threadIdx.x < st) red[threadIdx.x] += red[threadIdx.x + st];
        __syncthreads();
    }
    if (threadIdx.x == 0) mean[ba] = red[0] * (1.0f / NPTS);
}

__global__ void k_max_norm(const float* __restrict__ coords, const float* __restrict__ mean,
                           float* __restrict__ maxn) {
    int b = blockIdx.x;
    float m0 = mean[b*3+0], m1 = mean[b*3+1], m2 = mean[b*3+2];
    const float* c0 = coords + (size_t)(b*3+0) * NPTS;
    const float* c1 = coords + (size_t)(b*3+1) * NPTS;
    const float* c2 = coords + (size_t)(b*3+2) * NPTS;
    __shared__ float red[256];
    float mx = 0.f;
    for (int i = threadIdx.x; i < NPTS; i += 256) {
        float x = c0[i]-m0, y = c1[i]-m1, z = c2[i]-m2;
        mx = fmaxf(mx, sqrtf(x*x + y*y + z*z));
    }
    red[threadIdx.x] = mx; __syncthreads();
    for (int st = 128; st > 0; st >>= 1) {
        if (threadIdx.x < st) red[threadIdx.x] = fmaxf(red[threadIdx.x], red[threadIdx.x+st]);
        __syncthreads();
    }
    if (threadIdx.x == 0) maxn[b] = red[0];
}

__global__ void k_assign(const float* __restrict__ coords, const float* __restrict__ mean,
                         const float* __restrict__ maxn, float* __restrict__ nc,
                         int* __restrict__ vidx, float* __restrict__ cnt) {
    int idx = blockIdx.x * 256 + threadIdx.x;        // b*NPTS + n
    int b = idx >> 15, n = idx & (NPTS - 1);
    float scale = 1.0f / (maxn[b] * 2.0f);
    int vi[3];
    #pragma unroll
    for (int a = 0; a < 3; ++a) {
        float v = (coords[(size_t)(b*3+a)*NPTS + n] - mean[b*3+a]) * scale + 0.5f;
        v = fminf(fmaxf(v * (float)RES, 0.0f), (float)(RES - 1));
        nc[(size_t)(b*3+a)*NPTS + n] = v;
        vi[a] = __float2int_rn(v);                   // round-half-even, matches jnp.round
    }
    int flat = vi[0]*RES*RES + vi[1]*RES + vi[2];
    vidx[idx] = flat;
    atomicAdd(cnt + ((size_t)b << 15) + flat, 1.0f);
}

__global__ void k_scatter(const float* __restrict__ features, const int* __restrict__ vidx,
                          float* __restrict__ gsum) {
    int idx = blockIdx.x * 256 + threadIdx.x;        // bc*NPTS + n
    int bc = idx >> 15, n = idx & (NPTS - 1);
    int b  = bc >> 6;
    int fl = vidx[((size_t)b << 15) + n];
    atomicAdd(gsum + ((size_t)bc << 15) + fl, features[(size_t)bc * NPTS + n]);
}

__global__ void k_avg_to_f16(const float* __restrict__ gsum, const float* __restrict__ cnt,
                             _Float16* __restrict__ grid16) {
    int idx = blockIdx.x * 256 + threadIdx.x;        // ((b*RC)+s)*64 + c
    int c = idx & 63, s = (idx >> 6) & (RC - 1), b = idx >> 21;
    float cv = fmaxf(cnt[((size_t)b << 15) + s], 1.0f);
    float g  = gsum[(((size_t)(b*64 + c)) << 15) + s] / cv;
    grid16[idx] = (_Float16)g;
}

__global__ void k_wprep(const float* __restrict__ w, _Float16* __restrict__ wf) {
    int idx = blockIdx.x * 256 + threadIdx.x;        // tap*4096 + co*64 + ci
    int ci = idx & 63, co = (idx >> 6) & 63, tap = idx >> 12;
    wf[idx] = (_Float16)w[(size_t)(co*64 + ci) * 27 + tap];
}

__global__ void k_f32_to_f16(const float* __restrict__ x, _Float16* __restrict__ y, int n) {
    int i = blockIdx.x * 256 + threadIdx.x;
    if (i < n) y[i] = (_Float16)x[i];
}

__global__ void k_featprep(const float* __restrict__ f, _Float16* __restrict__ y) {
    int idx = blockIdx.x * 256 + threadIdx.x;        // ((b*NPTS)+n)*64 + c
    int c = idx & 63, n = (idx >> 6) & (NPTS - 1), b = idx >> 21;
    y[idx] = (_Float16)f[(size_t)(b*64 + c) * NPTS + n];
}

// ------------------------------------------------ 3x3x3 conv as implicit-GEMM WMMA
// Double-buffered LDS, staged via CDNA5 async global->LDS, one barrier per K-step.
// in:  f16 spatial-major [b][32768][64]   wt: f16 [27][64][64]   out: f32 [b][64][32768]
__global__ void k_conv3x3_wmma(const _Float16* __restrict__ in, const _Float16* __restrict__ wt,
                               const float* __restrict__ bias, float* __restrict__ out) {
    __shared__ __align__(16) _Float16 At[2][64 * 32];
    __shared__ __align__(16) _Float16 Bt[2][128 * 32];
    const int tid = threadIdx.x, lane = tid & 31, wave = tid >> 5;
    const int b  = blockIdx.y;
    const int s0 = blockIdx.x * 128;
    const int z  = s0 >> 10;
    const int y0 = (s0 >> 5) & 31;
    const int m0 = (wave & 3) * 16;                  // channel tile
    const int n0 = (wave >> 2) * 64;                 // spatial tile base
    const _Float16* inB = in + ((size_t)b << 15) * 64;

    // stage K-step 'it' (it = tap*2 + kchunk) into LDS buffer 'buf'
    auto stage = [&](int it, int buf) {
        const int tap = it >> 1;
        const int kc  = (it & 1) << 5;
        const int di = C_DI[tap], dj = C_DJ[tap], dk = C_DK[tap];
        const int zi = z + di;
        {   // A tile 64x32: 256 lanes x 16B
            int row = tid >> 2, ch8 = (tid & 3) * 8;
            async_copy_b128(lds_off(&At[buf][row * 32 + ch8]),
                            wt + (size_t)tap * 4096 + row * 64 + kc + ch8);
        }
        // B tile 128x32 (im2col, SAME zero-pad): 512 chunks x 16B
        for (int cid = tid; cid < 512; cid += 256) {
            int p = cid >> 2, c8 = (cid & 3) * 8;
            int yy = y0 + (p >> 5) + dj;
            int xx = (p & 31) + dk;
            _Float16* dst = &Bt[buf][p * 32 + c8];
            if ((unsigned)zi < 32u && (unsigned)yy < 32u && (unsigned)xx < 32u) {
                async_copy_b128(lds_off(dst),
                                inB + (size_t)(((zi << 10) + (yy << 5) + xx) * 64 + kc + c8));
            } else {
                uint4 zz = {0u, 0u, 0u, 0u};
                *(uint4*)dst = zz;                    // ds_store zero fill
            }
        }
    };

    v8f acc[4] = {v8f{}, v8f{}, v8f{}, v8f{}};
    stage(0, 0);

    for (int it = 0; it < 54; ++it) {
        const int cur = it & 1;
        wait_async0();                                // own async writes done
        __syncthreads();                              // everyone's writes visible
        if (it + 1 < 54) stage(it + 1, cur ^ 1);      // overlap next stage with WMMA

        const _Float16* Ab = At[cur];
        const _Float16* Bb = Bt[cur];
        // A fragment (16x32 f16): lane<16 holds K0-7,16-23 ; lane>=16 K8-15,24-31
        F16x16 a;
        {
            int r  = (m0 + (lane & 15)) * 32;
            int ka = (lane < 16) ? 0 : 8;
            a.u[0] = *(const uint4*)&Ab[r + ka];
            a.u[1] = *(const uint4*)&Ab[r + ka + 16];
        }
        #pragma unroll
        for (int j = 0; j < 4; ++j) {
            int n  = (n0 + j * 16 + (lane & 15)) * 32;
            int ks = (lane < 16) ? 0 : 8;
            F16x16 bf;
            bf.u[0] = *(const uint4*)&Bb[n + ks];
            bf.u[1] = *(const uint4*)&Bb[n + ks + 16];
            acc[j] = __builtin_amdgcn_wmma_f32_16x16x32_f16(
                        false, a.v, false, bf.v, (short)0, acc[j], false, false);
        }
    }
    // ---- store D (+bias), f32 channel-major
    float* outB = out + (((size_t)b * 64) << 15);
    const int ncol = lane & 15;
    const int mhi  = (lane < 16) ? 0 : 8;
    #pragma unroll
    for (int j = 0; j < 4; ++j) {
        int scol = s0 + n0 + j * 16 + ncol;
        #pragma unroll
        for (int r = 0; r < 8; ++r) {
            int m = m0 + r + mhi;
            outB[((size_t)m << 15) + scol] = acc[j][r] + bias[m];
        }
    }
}

// ------------------------------------------------ point GEMM (M=64,K=64) via WMMA
// in: f16 [b][N][64]  wt: f16 [64co][64ci]  out: f32 [b][64][N]
__global__ void k_pointgemm_wmma(const _Float16* __restrict__ in, const _Float16* __restrict__ wt,
                                 const float* __restrict__ bias, float* __restrict__ out) {
    __shared__ __align__(16) _Float16 At[2][64 * 32];
    __shared__ __align__(16) _Float16 Bt[2][128 * 32];
    const int tid = threadIdx.x, lane = tid & 31, wave = tid >> 5;
    const int b   = blockIdx.y;
    const int n0g = blockIdx.x * 128;
    const int m0  = (wave & 3) * 16;
    const int n0  = (wave >> 2) * 64;
    const _Float16* inB = in + ((size_t)b * NPTS) * 64;

    auto stage = [&](int it, int buf) {
        const int kc = it << 5;
        {
            int row = tid >> 2, ch8 = (tid & 3) * 8;
            async_copy_b128(lds_off(&At[buf][row * 32 + ch8]),
                            wt + row * 64 + kc + ch8);
        }
        for (int cid = tid; cid < 512; cid += 256) {
            int p = cid >> 2, c8 = (cid & 3) * 8;
            async_copy_b128(lds_off(&Bt[buf][p * 32 + c8]),
                            inB + (size_t)(n0g + p) * 64 + kc + c8);
        }
    };

    v8f acc[4] = {v8f{}, v8f{}, v8f{}, v8f{}};
    stage(0, 0);

    #pragma unroll
    for (int it = 0; it < 2; ++it) {
        const int cur = it & 1;
        wait_async0();
        __syncthreads();
        if (it + 1 < 2) stage(it + 1, cur ^ 1);

        const _Float16* Ab = At[cur];
        const _Float16* Bb = Bt[cur];
        F16x16 a;
        {
            int r  = (m0 + (lane & 15)) * 32;
            int ka = (lane < 16) ? 0 : 8;
            a.u[0] = *(const uint4*)&Ab[r + ka];
            a.u[1] = *(const uint4*)&Ab[r + ka + 16];
        }
        #pragma unroll
        for (int j = 0; j < 4; ++j) {
            int n  = (n0 + j * 16 + (lane & 15)) * 32;
            int ks = (lane < 16) ? 0 : 8;
            F16x16 bf;
            bf.u[0] = *(const uint4*)&Bb[n + ks];
            bf.u[1] = *(const uint4*)&Bb[n + ks + 16];
            acc[j] = __builtin_amdgcn_wmma_f32_16x16x32_f16(
                        false, a.v, false, bf.v, (short)0, acc[j], false, false);
        }
    }
    float* outB = out + (((size_t)b * 64) << 15);
    const int ncol = lane & 15;
    const int mhi  = (lane < 16) ? 0 : 8;
    #pragma unroll
    for (int j = 0; j < 4; ++j) {
        int ncolg = n0g + n0 + j * 16 + ncol;
        #pragma unroll
        for (int r = 0; r < 8; ++r) {
            int m = m0 + r + mhi;
            outB[((size_t)m << 15) + ncolg] = acc[j][r] + bias[m];
        }
    }
}

// ------------------------------------------------ BN helpers
__global__ void k_chan_stats(const float* __restrict__ x, float* __restrict__ stats, int S) {
    int c = blockIdx.x;
    __shared__ float rs[256], rq[256];
    float s = 0.f, q = 0.f;
    for (int b = 0; b < BATCH; ++b) {
        const float* src = x + (size_t)(b * 64 + c) * S;
        for (int i = threadIdx.x; i < S; i += 256) {
            float v = src[i]; s += v; q += v * v;
        }
    }
    rs[threadIdx.x] = s; rq[threadIdx.x] = q; __syncthreads();
    for (int st = 128; st > 0; st >>= 1) {
        if (threadIdx.x < st) { rs[threadIdx.x] += rs[threadIdx.x+st]; rq[threadIdx.x] += rq[threadIdx.x+st]; }
        __syncthreads();
    }
    if (threadIdx.x == 0) { stats[2*c] = rs[0]; stats[2*c+1] = rq[0]; }
}

__global__ void k_bn_leaky_to_f16(const float* __restrict__ x, const float* __restrict__ stats,
                                  const float* __restrict__ g, const float* __restrict__ be,
                                  _Float16* __restrict__ y, float invcnt) {
    int idx = blockIdx.x * 256 + threadIdx.x;
    int c = idx & 63, s = (idx >> 6) & (RC - 1), b = idx >> 21;
    float mu  = stats[2*c] * invcnt;
    float var = stats[2*c+1] * invcnt - mu * mu;
    float v = (x[(size_t)(b*64 + c) * RC + s] - mu) * rsqrtf(var + BN_EPS) * g[c] + be[c];
    v = (v < 0.f) ? 0.1f * v : v;
    y[idx] = (_Float16)v;
}

__global__ void k_bn_leaky_inplace(float* __restrict__ x, const float* __restrict__ stats,
                                   const float* __restrict__ g, const float* __restrict__ be,
                                   float invcnt) {
    int idx = blockIdx.x * 256 + threadIdx.x;        // (b*64+c)*RC + s
    int c = (idx >> 15) & 63;
    float mu  = stats[2*c] * invcnt;
    float var = stats[2*c+1] * invcnt - mu * mu;
    float v = (x[idx] - mu) * rsqrtf(var + BN_EPS) * g[c] + be[c];
    x[idx] = (v < 0.f) ? 0.1f * v : v;
}

// trilinear devoxelize + point-branch BN+ReLU, accumulate into d_out
__global__ void k_devox_add(const float* __restrict__ grid, const float* __restrict__ nc,
                            const float* __restrict__ statsP, const float* __restrict__ gp,
                            const float* __restrict__ bep, float* __restrict__ out, float invcnt) {
    int tid = threadIdx.x;
    int pid = blockIdx.x * 4 + (tid >> 6);           // b*NPTS + n
    int c = tid & 63;
    int b = pid >> 15, n = pid & (NPTS - 1);
    float vx = nc[(size_t)(b*3+0)*NPTS + n];
    float vy = nc[(size_t)(b*3+1)*NPTS + n];
    float vz = nc[(size_t)(b*3+2)*NPTS + n];
    float fx = vx - floorf(vx), fy = vy - floorf(vy), fz = vz - floorf(vz);
    int lx = min(max((int)floorf(vx), 0), RES-1), hx = min(lx + 1, RES-1);
    int ly = min(max((int)floorf(vy), 0), RES-1), hy = min(ly + 1, RES-1);
    int lz = min(max((int)floorf(vz), 0), RES-1), hz = min(lz + 1, RES-1);
    const float* gb = grid + (((size_t)(b*64 + c)) << 15);
    float acc = 0.f;
    #pragma unroll
    for (int dx = 0; dx < 2; ++dx)
      #pragma unroll
      for (int dy = 0; dy < 2; ++dy)
        #pragma unroll
        for (int dz = 0; dz < 2; ++dz) {
            int ix = dx ? hx : lx, iy = dy ? hy : ly, iz = dz ? hz : lz;
            float w = (dx ? fx : 1.f - fx) * (dy ? fy : 1.f - fy) * (dz ? fz : 1.f - fz);
            acc += w * gb[ix*RES*RES + iy*RES + iz];
        }
    size_t oidx = (size_t)(b*64 + c) * NPTS + n;
    float mu  = statsP[2*c] * invcnt;
    float var = statsP[2*c+1] * invcnt - mu * mu;
    float p = (out[oidx] - mu) * rsqrtf(var + BN_EPS) * gp[c] + bep[c];
    p = fmaxf(p, 0.f);
    out[oidx] = acc + p;
}

// ---------------------------------------------------------------- launcher
extern "C" void kernel_launch(void* const* d_in, const int* in_sizes, int n_in,
                              void* d_out, int out_size, void* d_ws, size_t ws_size,
                              hipStream_t stream) {
    const float* features = (const float*)d_in[0];
    const float* coords   = (const float*)d_in[1];
    const float* w1 = (const float*)d_in[2];  const float* b1 = (const float*)d_in[3];
    const float* g1 = (const float*)d_in[4];  const float* be1= (const float*)d_in[5];
    const float* w2 = (const float*)d_in[6];  const float* b2 = (const float*)d_in[7];
    const float* g2 = (const float*)d_in[8];  const float* be2= (const float*)d_in[9];
    const float* wp = (const float*)d_in[10]; const float* bp = (const float*)d_in[11];
    const float* gp = (const float*)d_in[12]; const float* bep= (const float*)d_in[13];
    float* out = (float*)d_out;

    char* ws = (char*)d_ws; size_t off = 0;
    auto alloc = [&](size_t bytes) { void* p = ws + off; off = (off + bytes + 255) & ~(size_t)255; return p; };
    float*    mean   = (float*)alloc(BATCH*3*sizeof(float));
    float*    maxn   = (float*)alloc(BATCH*sizeof(float));
    float*    stats1 = (float*)alloc(64*2*sizeof(float));
    float*    stats2 = (float*)alloc(64*2*sizeof(float));
    float*    statsP = (float*)alloc(64*2*sizeof(float));
    _Float16* wf1    = (_Float16*)alloc((size_t)27*64*64*2);
    _Float16* wf2    = (_Float16*)alloc((size_t)27*64*64*2);
    _Float16* wpf    = (_Float16*)alloc((size_t)64*64*2);
    float*    ncbuf  = (float*)alloc((size_t)BATCH*3*NPTS*4);
    int*      vidx   = (int*)alloc((size_t)BATCH*NPTS*4);
    float*    cnt    = (float*)alloc((size_t)BATCH*RC*4);
    float*    W0     = (float*)alloc((size_t)BATCH*64*RC*4);   // gsum -> conv1 -> conv2 (reused)
    _Float16* H16    = (_Float16*)alloc((size_t)BATCH*RC*64*2);// grid16 -> act1 f16 (reused)
    _Float16* feat16 = (_Float16*)alloc((size_t)BATCH*NPTS*64*2);

    const int nGrid = BATCH*64*RC;                 // 16.7M
    const float invBN = 1.0f / (float)(BATCH * RC);

    // ---- voxelize
    k_zero_f32<<<(nGrid+255)/256, 256, 0, stream>>>(W0, nGrid);
    k_zero_f32<<<(BATCH*RC+255)/256, 256, 0, stream>>>(cnt, BATCH*RC);
    k_coord_mean<<<BATCH*3, 256, 0, stream>>>(coords, mean);
    k_max_norm<<<BATCH, 256, 0, stream>>>(coords, mean, maxn);
    k_assign<<<BATCH*NPTS/256, 256, 0, stream>>>(coords, mean, maxn, ncbuf, vidx, cnt);
    k_scatter<<<BATCH*64*NPTS/256, 256, 0, stream>>>(features, vidx, W0);
    k_avg_to_f16<<<nGrid/256, 256, 0, stream>>>(W0, cnt, H16);

    // ---- weight / feature repack
    k_wprep<<<27*64*64/256, 256, 0, stream>>>(w1, wf1);
    k_wprep<<<27*64*64/256, 256, 0, stream>>>(w2, wf2);
    k_f32_to_f16<<<(64*64+255)/256, 256, 0, stream>>>(wp, wpf, 64*64);
    k_featprep<<<BATCH*NPTS*64/256, 256, 0, stream>>>(features, feat16);

    // ---- conv1 -> BN stats -> act + repack f16
    k_conv3x3_wmma<<<dim3(RC/128, BATCH), 256, 0, stream>>>(H16, wf1, b1, W0);
    k_chan_stats<<<64, 256, 0, stream>>>(W0, stats1, RC);
    k_bn_leaky_to_f16<<<nGrid/256, 256, 0, stream>>>(W0, stats1, g1, be1, H16, invBN);

    // ---- conv2 -> BN stats -> act in place
    k_conv3x3_wmma<<<dim3(RC/128, BATCH), 256, 0, stream>>>(H16, wf2, b2, W0);
    k_chan_stats<<<64, 256, 0, stream>>>(W0, stats2, RC);
    k_bn_leaky_inplace<<<nGrid/256, 256, 0, stream>>>(W0, stats2, g2, be2, invBN);

    // ---- point branch GEMM into d_out, stats, then devox + add
    k_pointgemm_wmma<<<dim3(NPTS/128, BATCH), 256, 0, stream>>>(feat16, wpf, bp, out);
    k_chan_stats<<<64, 256, 0, stream>>>(out, statsP, NPTS);
    k_devox_add<<<BATCH*NPTS/4, 256, 0, stream>>>(W0, ncbuf, statsP, gp, bep, out, invBN);
}
